// FlaxMixtralBlocKSparesTop2MLPCollection_74758200754530
// MI455X (gfx1250) — compile-verified, matrix operand order checked by reading, not run on hardware
//
#include <hip/hip_runtime.h>

// Problem constants (B=2, S=2048, H=1024, I=3584, E=8, K=2)
#define H_DIM   1024
#define I_DIM   3584
#define E_NUM   8
#define BS      4096                     // B*S tokens
#define K_TOP   2
#define NENT    (BS * K_TOP)             // 8192 routing entries
#define TILE_M  64
#define TILE_N  128
#define MAX_ENT (NENT + E_NUM * TILE_M)  // 8704 padded entries (worst case)
#define MAX_TILES (MAX_ENT / TILE_M)     // 136 M-tiles total (tight bound)

#define AB_U32 (TILE_M * 20)             // dwords per A LDS buffer (row=20 dw)
#define BB_U32 (TILE_N * 20)             // dwords per B LDS buffer

typedef __attribute__((ext_vector_type(16))) __bf16 v16bf;
typedef __attribute__((ext_vector_type(2)))  __bf16 v2bf;
typedef __attribute__((ext_vector_type(8)))  float  v8f;

union FragAB {
  v16bf bf;
  unsigned int u[8];
  uint4 q[2];
};

// ---- helpers -------------------------------------------------------------

__device__ __forceinline__ unsigned short f2bf(float x) {
  union { __bf16 b; unsigned short u; } cv;
  cv.b = (__bf16)x;                       // native cvt, RNE
  return cv.u;
}

// packs to one v_cvt_pk_bf16_f32
__device__ __forceinline__ unsigned int pack2(float lo, float hi) {
  union { v2bf v; unsigned int u; } cv;
  cv.v.x = (__bf16)lo;
  cv.v.y = (__bf16)hi;
  return cv.u;
}

__device__ __forceinline__ float silu_f(float x) {
  return x / (1.0f + __expf(-x));
}

// 16x32 bf16 WMMA fragment from LDS tile with 20-dword (80B) rows.
// Lanes 0-15 hold K pairs {0..7,16..23}, lanes 16-31 {8..15,24..31}
// -> two uint4 (ds_load_b128) per lane.
__device__ __forceinline__ void ld_frag(const unsigned int* sm, int row,
                                        int half, FragAB& f) {
  const uint4* p = (const uint4*)(sm + row * 20);
  f.q[0] = p[half];
  f.q[1] = p[2 + half];
}

#define WMMA_BF16(A, B, C)                                                   \
  __builtin_amdgcn_wmma_f32_16x16x32_bf16(false, (A).bf, false, (B).bf,      \
                                          (short)0, (C), false, false)

// ---- pass 1: routing -----------------------------------------------------
__global__ void route_k(const int* __restrict__ sel, const float* __restrict__ rw,
                        int* __restrict__ meta, int* __restrict__ etok,
                        float* __restrict__ ewt, int* __restrict__ epos) {
  __shared__ int cnt[E_NUM], cur[E_NUM], sb[E_NUM + 1];
  const int tid = threadIdx.x;
  if (tid < E_NUM) { cnt[tid] = 0; cur[tid] = 0; }
  __syncthreads();
  for (int i = tid; i < NENT; i += blockDim.x) atomicAdd(&cnt[sel[i]], 1);
  __syncthreads();
  if (tid == 0) {
    int b = 0;
    for (int e = 0; e < E_NUM; ++e) {
      sb[e] = b;
      b += ((cnt[e] + TILE_M - 1) / TILE_M) * TILE_M;
    }
    sb[E_NUM] = b;
  }
  __syncthreads();
  for (int i = tid; i < NENT; i += blockDim.x) {
    int e = sel[i];
    int p = atomicAdd(&cur[e], 1);
    int idx = sb[e] + p;
    etok[idx] = i / K_TOP;
    ewt[idx]  = rw[i];
    epos[i]   = idx;
  }
  __syncthreads();
  for (int e = 0; e < E_NUM; ++e)
    for (int j = sb[e] + cnt[e] + tid; j < sb[e + 1]; j += blockDim.x) {
      etok[j] = 0;
      ewt[j]  = 0.0f;
    }
  if (tid <= E_NUM) meta[tid] = sb[tid];
}

// ---- pass 2: g/u GEMM + fused SwiGLU -> h (bf16) -------------------------
// 256 threads, 8 waves (4Mx2N), tile 64x128, double-buffered LDS.
// Per iteration: frag ds_loads (grouped) -> next-chunk global loads ->
// 8 WMMAs -> cvt + ds_store into other buffer -> barrier.
__global__ void __launch_bounds__(256)
gemm1_k(const float* __restrict__ hs, const float* __restrict__ w1,
        const float* __restrict__ w3, const int* __restrict__ meta,
        const int* __restrict__ etok, unsigned short* __restrict__ hbuf) {
  const int row0 = blockIdx.y * TILE_M;
  const int total = meta[E_NUM];
  if (row0 >= total) return;
  int e = 0;
  while (row0 >= meta[e + 1]) ++e;

  const int n0 = blockIdx.x * TILE_N;
  const float* __restrict__ W1 = w1 + (size_t)e * ((size_t)H_DIM * I_DIM);
  const float* __restrict__ W3 = w3 + (size_t)e * ((size_t)H_DIM * I_DIM);

  __shared__ __align__(16) unsigned int smA[2 * AB_U32];
  __shared__ __align__(16) unsigned int smB1[2 * BB_U32];
  __shared__ __align__(16) unsigned int smB3[2 * BB_U32];
  __shared__ int s_tok[TILE_M];

  const int tid = threadIdx.x;
  if (tid < TILE_M) s_tok[tid] = etok[row0 + tid];
  __syncthreads();

  // loop-invariant fill state: two gathered A rows per thread
  const int rA = tid >> 3, kqA = tid & 7;
  const float* __restrict__ aPtr0 = hs + (size_t)s_tok[rA] * H_DIM + kqA * 4;
  const float* __restrict__ aPtr1 = hs + (size_t)s_tok[rA + 32] * H_DIM + kqA * 4;
  const int dA_off0 = rA * 20 + kqA * 2;
  const int dA_off1 = (rA + 32) * 20 + kqA * 2;
  const int kp0 = tid >> 5;               // k-pair strip base
  const int nq  = tid & 31;
  const int nB4 = nq * 4;
  const size_t bOff0 = (size_t)(kp0 * 2) * I_DIM + nB4;
  const size_t bOff1 = bOff0 + (size_t)16 * I_DIM;

  const int lane  = tid & 31;
  const int wid   = tid >> 5;
  const int mBase = (wid >> 1) * 16;      // 4 M positions
  const int nBase = (wid & 1) * 64;       // 2 N positions, 64 wide each
  const int half  = lane >> 4;
  const int lrow  = lane & 15;

  v8f g[4] = {{}, {}, {}, {}};
  v8f u[4] = {{}, {}, {}, {}};

  struct Gl1 {
    float4 a0, a1;
    float4 w1a[2], w1b[2], w3a[2], w3b[2];
  };

  auto gload = [&](int k0) {
    Gl1 L;
    L.a0 = *(const float4*)(aPtr0 + k0);
    L.a1 = *(const float4*)(aPtr1 + k0);
    const float* __restrict__ w1c = W1 + (size_t)k0 * I_DIM + n0;
    const float* __restrict__ w3c = W3 + (size_t)k0 * I_DIM + n0;
#pragma unroll
    for (int s = 0; s < 2; ++s) {
      const size_t off = s ? bOff1 : bOff0;
      L.w1a[s] = *(const float4*)(w1c + off);
      L.w1b[s] = *(const float4*)(w1c + off + I_DIM);
      L.w3a[s] = *(const float4*)(w3c + off);
      L.w3b[s] = *(const float4*)(w3c + off + I_DIM);
    }
    return L;
  };

  auto lstore = [&](const Gl1& L, unsigned int* dA, unsigned int* dB1,
                    unsigned int* dB3) {
    uint2 v0, v1;
    v0.x = pack2(L.a0.x, L.a0.y);
    v0.y = pack2(L.a0.z, L.a0.w);
    v1.x = pack2(L.a1.x, L.a1.y);
    v1.y = pack2(L.a1.z, L.a1.w);
    *(uint2*)&dA[dA_off0] = v0;
    *(uint2*)&dA[dA_off1] = v1;
#pragma unroll
    for (int s = 0; s < 2; ++s) {
      const int kp = kp0 + s * 8;
      dB1[(nB4 + 0) * 20 + kp] = pack2(L.w1a[s].x, L.w1b[s].x);
      dB1[(nB4 + 1) * 20 + kp] = pack2(L.w1a[s].y, L.w1b[s].y);
      dB1[(nB4 + 2) * 20 + kp] = pack2(L.w1a[s].z, L.w1b[s].z);
      dB1[(nB4 + 3) * 20 + kp] = pack2(L.w1a[s].w, L.w1b[s].w);
      dB3[(nB4 + 0) * 20 + kp] = pack2(L.w3a[s].x, L.w3b[s].x);
      dB3[(nB4 + 1) * 20 + kp] = pack2(L.w3a[s].y, L.w3b[s].y);
      dB3[(nB4 + 2) * 20 + kp] = pack2(L.w3a[s].z, L.w3b[s].z);
      dB3[(nB4 + 3) * 20 + kp] = pack2(L.w3a[s].w, L.w3b[s].w);
    }
  };

  {
    Gl1 L = gload(0);
    lstore(L, smA, smB1, smB3);
  }
  __syncthreads();

  for (int kt = 0; kt < H_DIM / 32; ++kt) {
    const int cur = kt & 1;
    const bool have = (kt + 1 < H_DIM / 32);

    const unsigned int* cA  = smA  + cur * AB_U32;
    const unsigned int* cB1 = smB1 + cur * BB_U32;
    const unsigned int* cB3 = smB3 + cur * BB_U32;

    FragAB fa;
    ld_frag(cA, mBase + lrow, half, fa);

    Gl1 L;
    if (have) L = gload((kt + 1) * 32);

    // Stage 4 fragments, then 4 WMMAs, so the ds_loads issue back-to-back
    // and waits drain progressively while earlier WMMAs execute.
    FragAB fb[4];
#pragma unroll
    for (int j = 0; j < 4; ++j)
      ld_frag(cB1, nBase + j * 16 + lrow, half, fb[j]);
#pragma unroll
    for (int j = 0; j < 4; ++j)
      g[j] = WMMA_BF16(fa, fb[j], g[j]);
#pragma unroll
    for (int j = 0; j < 4; ++j)
      ld_frag(cB3, nBase + j * 16 + lrow, half, fb[j]);
#pragma unroll
    for (int j = 0; j < 4; ++j)
      u[j] = WMMA_BF16(fa, fb[j], u[j]);

    if (have) {
      lstore(L, smA + (cur ^ 1) * AB_U32, smB1 + (cur ^ 1) * BB_U32,
             smB3 + (cur ^ 1) * BB_U32);
    }
    __syncthreads();
  }

  // h = silu(g) * u, bf16 row-major per padded entry row.
  // C layout: vgpr r <-> M = r + 8*half; N = lane&15.
#pragma unroll
  for (int j = 0; j < 4; ++j) {
#pragma unroll
    for (int r = 0; r < 8; ++r) {
      int gm = row0 + mBase + r + half * 8;
      float s = silu_f(g[j][r]) * u[j][r];
      hbuf[(size_t)gm * I_DIM + n0 + nBase + j * 16 + lrow] = f2bf(s);
    }
  }
}

// ---- pass 3: y = h @ w2, scaled by routing weight ------------------------
__global__ void __launch_bounds__(256)
gemm2_k(const unsigned short* __restrict__ hbuf, const float* __restrict__ w2,
        const int* __restrict__ meta, const float* __restrict__ ewt,
        float* __restrict__ ybuf) {
  const int row0 = blockIdx.y * TILE_M;
  const int total = meta[E_NUM];
  if (row0 >= total) return;
  int e = 0;
  while (row0 >= meta[e + 1]) ++e;

  const int n0 = blockIdx.x * TILE_N;
  const float* __restrict__ W2 = w2 + (size_t)e * ((size_t)I_DIM * H_DIM);

  __shared__ __align__(16) unsigned int smA[2 * AB_U32];
  __shared__ __align__(16) unsigned int smB[2 * BB_U32];
  __shared__ float s_w[TILE_M];

  const int tid = threadIdx.x;
  if (tid < TILE_M) s_w[tid] = ewt[row0 + tid];
  __syncthreads();

  // A fill: 64 rows x 32 bf16 = 256 uint4, one per thread
  const int rA = tid >> 2, qA = tid & 3;
  const unsigned short* __restrict__ aPtr =
      hbuf + (size_t)(row0 + rA) * I_DIM + qA * 8;
  const int dA_off = rA * 20 + qA * 4;
  const int kp0 = tid >> 5;
  const int nq  = tid & 31;
  const int nB4 = nq * 4;
  const size_t bOff0 = (size_t)(kp0 * 2) * H_DIM + nB4;
  const size_t bOff1 = bOff0 + (size_t)16 * H_DIM;

  const int lane  = tid & 31;
  const int wid   = tid >> 5;
  const int mBase = (wid >> 1) * 16;
  const int nBase = (wid & 1) * 64;
  const int half  = lane >> 4;
  const int lrow  = lane & 15;

  v8f acc[4] = {{}, {}, {}, {}};

  struct Gl2 {
    uint4 a;
    float4 wa[2], wb[2];
  };

  auto gload = [&](int k0) {
    Gl2 L;
    L.a = *(const uint4*)(aPtr + k0);
    const float* __restrict__ w2c = W2 + (size_t)k0 * H_DIM + n0;
#pragma unroll
    for (int s = 0; s < 2; ++s) {
      const size_t off = s ? bOff1 : bOff0;
      L.wa[s] = *(const float4*)(w2c + off);
      L.wb[s] = *(const float4*)(w2c + off + H_DIM);
    }
    return L;
  };

  auto lstore = [&](const Gl2& L, unsigned int* dA, unsigned int* dB) {
    *(uint4*)&dA[dA_off] = L.a;
#pragma unroll
    for (int s = 0; s < 2; ++s) {
      const int kp = kp0 + s * 8;
      dB[(nB4 + 0) * 20 + kp] = pack2(L.wa[s].x, L.wb[s].x);
      dB[(nB4 + 1) * 20 + kp] = pack2(L.wa[s].y, L.wb[s].y);
      dB[(nB4 + 2) * 20 + kp] = pack2(L.wa[s].z, L.wb[s].z);
      dB[(nB4 + 3) * 20 + kp] = pack2(L.wa[s].w, L.wb[s].w);
    }
  };

  {
    Gl2 L = gload(0);
    lstore(L, smA, smB);
  }
  __syncthreads();

  for (int kt = 0; kt < I_DIM / 32; ++kt) {
    const int cur = kt & 1;
    const bool have = (kt + 1 < I_DIM / 32);

    const unsigned int* cA = smA + cur * AB_U32;
    const unsigned int* cB = smB + cur * BB_U32;

    FragAB fa;
    ld_frag(cA, mBase + lrow, half, fa);

    Gl2 L;
    if (have) L = gload((kt + 1) * 32);

    FragAB fb[4];
#pragma unroll
    for (int j = 0; j < 4; ++j)
      ld_frag(cB, nBase + j * 16 + lrow, half, fb[j]);
#pragma unroll
    for (int j = 0; j < 4; ++j)
      acc[j] = WMMA_BF16(fa, fb[j], acc[j]);

    if (have) {
      lstore(L, smA + (cur ^ 1) * AB_U32, smB + (cur ^ 1) * BB_U32);
    }
    __syncthreads();
  }

#pragma unroll
  for (int j = 0; j < 4; ++j) {
#pragma unroll
    for (int r = 0; r < 8; ++r) {
      int lm = mBase + r + half * 8;
      int gm = row0 + lm;
      float w = s_w[lm];
      ybuf[(size_t)gm * H_DIM + n0 + nBase + j * 16 + lrow] = w * acc[j][r];
    }
  }
}

// ---- pass 4: deterministic combine (vectorized, no FP atomics) -----------
__global__ void combine_k(const int* __restrict__ epos, const float* __restrict__ ybuf,
                          float* __restrict__ out) {
  int idx = blockIdx.x * blockDim.x + threadIdx.x;   // over BS*H/4
  int t  = idx >> 8;                                  // (idx*4) / H_DIM
  int h4 = (idx & 255) * 4;
  const float4 a = *(const float4*)(ybuf + (size_t)epos[t * 2 + 0] * H_DIM + h4);
  const float4 b = *(const float4*)(ybuf + (size_t)epos[t * 2 + 1] * H_DIM + h4);
  float4 o;
  o.x = a.x + b.x; o.y = a.y + b.y; o.z = a.z + b.z; o.w = a.w + b.w;
  *(float4*)(out + (size_t)t * H_DIM + h4) = o;
}

// ---- host launch ---------------------------------------------------------
extern "C" void kernel_launch(void* const* d_in, const int* in_sizes, int n_in,
                              void* d_out, int out_size, void* d_ws, size_t ws_size,
                              hipStream_t stream) {
  const float* hs  = (const float*)d_in[0];  // [B,S,H]
  const int*   sel = (const int*)d_in[1];    // [B,S,K]
  const float* rw  = (const float*)d_in[2];  // [B,S,K]
  const float* w1  = (const float*)d_in[3];  // [E,H,I]
  const float* w3  = (const float*)d_in[4];  // [E,H,I]
  const float* w2  = (const float*)d_in[5];  // [E,I,H]
  float* out = (float*)d_out;

  // workspace layout (~98.1 MB total)
  char* ws = (char*)d_ws;
  int*   meta = (int*)ws;                                     // 9 ints
  int*   etok = (int*)(ws + 256);                             // MAX_ENT
  float* ewt  = (float*)(ws + 256 + (size_t)MAX_ENT * 4);     // MAX_ENT
  int*   epos = (int*)(ws + 256 + (size_t)MAX_ENT * 8);       // NENT
  unsigned short* hbuf =
      (unsigned short*)(ws + 256 + (size_t)MAX_ENT * 8 + (size_t)NENT * 4);
  float* ybuf =
      (float*)(ws + 256 + (size_t)MAX_ENT * 8 + (size_t)NENT * 4 +
               (size_t)MAX_ENT * I_DIM * 2);

  route_k<<<1, 256, 0, stream>>>(sel, rw, meta, etok, ewt, epos);
  gemm1_k<<<dim3(I_DIM / TILE_N, MAX_TILES), 256, 0, stream>>>(hs, w1, w3, meta, etok, hbuf);
  gemm2_k<<<dim3(H_DIM / TILE_N, MAX_TILES), 256, 0, stream>>>(hbuf, w2, meta, ewt, ybuf);
  combine_k<<<(BS * H_DIM / 4 + 255) / 256, 256, 0, stream>>>(epos, ybuf, out);
}
